// MoELayer_84817014161795
// MI455X (gfx1250) — matrix operand order repaired and checked by reading, not compiled, not run on hardware
//
#include <hip/hip_runtime.h>
#include <hip/hip_bf16.h>
#include <stdint.h>

typedef __bf16 bf16;
typedef __attribute__((ext_vector_type(16))) __bf16 v16bf;
typedef __attribute__((ext_vector_type(8)))  float  v8f;
typedef __attribute__((ext_vector_type(8)))  unsigned short v8us;

#define D_MODEL 1024
#define D_FF    4096
#define N_EXP   8
#define T_TOK   4096
#define M_TILE  32          // tokens per block (two 16-row WMMA M-halves)
#define F_CHUNK 2048
#define NTHREADS 512        // 16 waves
#define XSTR    (D_MODEL + 8)   // halfword stride: 16B-aligned, breaks bank conflicts
#define HSTR    (F_CHUNK + 8)

// ---------------- fragment loaders ----------------
// A fragment (16x32 bf16, contiguous K order matching frag_b):
// lanes 0-15: M=lane, K = kbase+[0..15]
// lanes 16-31: M=lane-16, K = kbase+[16..31]
__device__ inline v16bf frag_a(const __bf16* lds16rows, int stride, int kbase) {
  int lane = threadIdx.x & 31;
  int row  = lane & 15;
  int c0   = kbase + ((lane >> 4) << 4);
  const __bf16* p = lds16rows + (size_t)row * stride + c0;
  union { v16bf v; v8us h[2]; } u;
  u.h[0] = *(const v8us*)(p);
  u.h[1] = *(const v8us*)(p + 8);
  return u.v;
}

// B fragment (32x16 bf16): weight tiles pre-swizzled so lane L's 16 values
// (K = (L>>4)*16 + 0..15 at column N = L&15) are contiguous (2 x b128).
__device__ inline v16bf frag_b(const __bf16* tile) {
  int lane = threadIdx.x & 31;
  const v8us* p = (const v8us*)tile + lane * 2;
  union { v16bf v; v8us h[2]; } u;
  u.h[0] = p[0];
  u.h[1] = p[1];
  return u.v;
}

// ---------------- conversion kernels ----------------
__global__ void cvt_x_kernel(const float* __restrict__ x, bf16* __restrict__ xb, int n) {
  for (int i = blockIdx.x * blockDim.x + threadIdx.x; i < n; i += gridDim.x * blockDim.x)
    xb[i] = (bf16)x[i];
}

// w: [E][K][N] row-major fp32  ->  wt: [E][N/16][K/32][(khalf*16+n)][16] bf16
__global__ void cvt_w_tiled_kernel(const float* __restrict__ w, bf16* __restrict__ wt,
                                   int K, int N) {
  size_t total = (size_t)N_EXP * K * N;
  for (size_t i = (size_t)blockIdx.x * blockDim.x + threadIdx.x; i < total;
       i += (size_t)gridDim.x * blockDim.x) {
    size_t e   = i / ((size_t)K * N);
    size_t rem = i - e * (size_t)K * N;
    int k = (int)(rem / N), n = (int)(rem % N);
    int kt = k >> 5, kin = k & 31, khalf = kin >> 4, kk = kin & 15;
    int nt = n >> 4, nn = n & 15;
    size_t o = ((((e * (N / 16) + nt) * (K / 32) + kt) * 32) + (khalf * 16 + nn)) * 16 + kk;
    wt[o] = (bf16)w[i];
  }
}

// ---------------- router ----------------
__global__ void router_kernel(const float* __restrict__ x, const float* __restrict__ Wr,
                              const float* __restrict__ br, int* __restrict__ counts,
                              int* __restrict__ etok, float* __restrict__ egate) {
  int t = blockIdx.x * blockDim.x + threadIdx.x;
  if (t >= T_TOK) return;
  float acc[N_EXP];
#pragma unroll
  for (int e = 0; e < N_EXP; ++e) acc[e] = br[e];
  const float* xr = x + (size_t)t * D_MODEL;
  for (int d = 0; d < D_MODEL; ++d) {
    float xv = xr[d];
    const float* wrow = Wr + d * N_EXP;
#pragma unroll
    for (int e = 0; e < N_EXP; ++e) acc[e] = fmaf(xv, wrow[e], acc[e]);
  }
  int i0 = 0;
  for (int e = 1; e < N_EXP; ++e) if (acc[e] > acc[i0]) i0 = e;
  int i1 = -1;
  for (int e = 0; e < N_EXP; ++e) {
    if (e == i0) continue;
    if (i1 < 0 || acc[e] > acc[i1]) i1 = e;
  }
  float m  = acc[i0];
  float p0 = __expf(acc[i0] - m), p1 = __expf(acc[i1] - m);
  float inv = 1.0f / (p0 + p1);
  int s0 = atomicAdd(&counts[i0], 1);
  etok[i0 * T_TOK + s0] = t; egate[i0 * T_TOK + s0] = p0 * inv;
  int s1 = atomicAdd(&counts[i1], 1);
  etok[i1 * T_TOK + s1] = t; egate[i1 * T_TOK + s1] = p1 * inv;
}

// ---------------- fused expert FFN ----------------
// Block: 512 threads = 16 waves, 32 tokens. Wave w: M-half = w&1, N-slice = w>>1.
__global__ __launch_bounds__(NTHREADS) void moe_ffn_kernel(
    const bf16* __restrict__ xb, const bf16* __restrict__ W1t, const bf16* __restrict__ W2t,
    const float* __restrict__ b1, const float* __restrict__ b2,
    const int* __restrict__ counts, const int* __restrict__ etok,
    const float* __restrict__ egate, float* __restrict__ out) {
  __shared__ alignas(16) __bf16 sX[M_TILE][XSTR];     // 66 KB
  __shared__ alignas(16) __bf16 sH[M_TILE][HSTR];     // 131.6 KB
  __shared__ int   sTok[M_TILE];
  __shared__ float sGate[M_TILE];

  const int e    = blockIdx.y;
  const int cnt  = counts[e];
  const int row0 = blockIdx.x * M_TILE;
  if (row0 >= cnt) return;

  const int tid   = threadIdx.x;
  const int lane  = tid & 31;
  const int wave  = tid >> 5;       // 0..15
  const int mhalf = wave & 1;       // which 16-row half of the 32-token tile
  const int nslc  = wave >> 1;      // 0..7 N-slice

  if (tid < M_TILE) {
    int gi = row0 + tid;
    sTok[tid]  = (gi < cnt) ? etok[e * T_TOK + gi] : -1;
    sGate[tid] = (gi < cnt) ? egate[e * T_TOK + gi] : 0.0f;
  }
  __syncthreads();

  // stage 32 x-rows (bf16) into LDS; 16 threads/row, 64 elems each
  {
    int r = tid >> 4, cbase = (tid & 15) * 64;
    int tok = sTok[r];
    if (tok >= 0) {
      const bf16* src = xb + (size_t)tok * D_MODEL + cbase;
#pragma unroll
      for (int j = 0; j < 8; ++j)
        *(v8us*)&sX[r][cbase + j * 8] = *(const v8us*)(src + j * 8);
    } else {
      v8us z = {};
#pragma unroll
      for (int j = 0; j < 8; ++j) *(v8us*)&sX[r][cbase + j * 8] = z;
    }
  }
  __syncthreads();

  const __bf16* aX = &sX[mhalf * 16][0];
  const __bf16* aH = &sH[mhalf * 16][0];
  const int m0 = (lane >> 4) * 8;
  const int nn = lane & 15;

  v8f yacc[8];
#pragma unroll
  for (int j = 0; j < 8; ++j) yacc[j] = (v8f){};

  for (int fc = 0; fc < D_FF / F_CHUNK; ++fc) {
    // ---- Phase 1: sH = relu(x @ W1[:, chunk] + b1)  (bf16 into LDS) ----
    // wave computes 16 n-tiles (nt = nslc*16 + 0..15) for its M-half,
    // in 4 groups of 4 so one A fragment feeds 4 WMMAs.
#pragma unroll 1
    for (int g = 0; g < 4; ++g) {
      int ntb = nslc * 16 + g * 4;               // first n-tile of group (chunk-local)
      const bf16* bt[4];
#pragma unroll
      for (int jj = 0; jj < 4; ++jj) {
        int gnt = fc * (F_CHUNK / 16) + ntb + jj;
        bt[jj] = W1t + (((size_t)e * (D_FF / 16) + gnt) * (D_MODEL / 32)) * 512;
      }
      v8f c0 = (v8f){}, c1 = (v8f){}, c2 = (v8f){}, c3 = (v8f){};
#pragma unroll 2
      for (int kt = 0; kt < D_MODEL / 32; ++kt) {
        v16bf a = frag_a(aX, XSTR, kt * 32);     // one A fragment, 4 uses
        size_t ko = (size_t)kt * 512;
        c0 = __builtin_amdgcn_wmma_f32_16x16x32_bf16(false, a, false, frag_b(bt[0] + ko),
                                                     (short)0, c0, false, false);
        c1 = __builtin_amdgcn_wmma_f32_16x16x32_bf16(false, a, false, frag_b(bt[1] + ko),
                                                     (short)0, c1, false, false);
        c2 = __builtin_amdgcn_wmma_f32_16x16x32_bf16(false, a, false, frag_b(bt[2] + ko),
                                                     (short)0, c2, false, false);
        c3 = __builtin_amdgcn_wmma_f32_16x16x32_bf16(false, a, false, frag_b(bt[3] + ko),
                                                     (short)0, c3, false, false);
      }
      v8f cs[4] = {c0, c1, c2, c3};
#pragma unroll
      for (int jj = 0; jj < 4; ++jj) {
        int nt  = ntb + jj;
        int gnt = fc * (F_CHUNK / 16) + nt;
        float bias = b1[(size_t)e * D_FF + gnt * 16 + nn];
#pragma unroll
        for (int r = 0; r < 8; ++r) {
          float v = cs[jj][r] + bias;
          v = v > 0.0f ? v : 0.0f;
          sH[mhalf * 16 + m0 + r][nt * 16 + nn] = (bf16)v;
        }
      }
    }
    __syncthreads();

    // ---- Phase 2: yacc += sH_chunk @ W2[chunk rows, :] ----
    // wave owns 8 D-tiles (nt2 = nslc*8 + 0..7) for its M-half; groups of 4.
#pragma unroll 1
    for (int g = 0; g < 2; ++g) {
      const bf16* bt[4];
#pragma unroll
      for (int jj = 0; jj < 4; ++jj) {
        int nt2 = nslc * 8 + g * 4 + jj;
        bt[jj] = W2t +
                 (((size_t)e * (D_MODEL / 16) + nt2) * (D_FF / 32) + fc * (F_CHUNK / 32)) * 512;
      }
      v8f c0 = yacc[g * 4 + 0], c1 = yacc[g * 4 + 1];
      v8f c2 = yacc[g * 4 + 2], c3 = yacc[g * 4 + 3];
#pragma unroll 2
      for (int kt = 0; kt < F_CHUNK / 32; ++kt) {
        v16bf a = frag_a(aH, HSTR, kt * 32);
        size_t ko = (size_t)kt * 512;
        c0 = __builtin_amdgcn_wmma_f32_16x16x32_bf16(false, a, false, frag_b(bt[0] + ko),
                                                     (short)0, c0, false, false);
        c1 = __builtin_amdgcn_wmma_f32_16x16x32_bf16(false, a, false, frag_b(bt[1] + ko),
                                                     (short)0, c1, false, false);
        c2 = __builtin_amdgcn_wmma_f32_16x16x32_bf16(false, a, false, frag_b(bt[2] + ko),
                                                     (short)0, c2, false, false);
        c3 = __builtin_amdgcn_wmma_f32_16x16x32_bf16(false, a, false, frag_b(bt[3] + ko),
                                                     (short)0, c3, false, false);
      }
      yacc[g * 4 + 0] = c0; yacc[g * 4 + 1] = c1;
      yacc[g * 4 + 2] = c2; yacc[g * 4 + 3] = c3;
    }
    __syncthreads();
  }

  // ---- epilogue: +b2, gate-scale, scatter-add into out ----
#pragma unroll 1
  for (int j = 0; j < 8; ++j) {
    int col = (nslc * 8 + j) * 16 + nn;
    float bias = b2[(size_t)e * D_MODEL + col];
#pragma unroll
    for (int r = 0; r < 8; ++r) {
      int m = mhalf * 16 + m0 + r;
      int tok = sTok[m];
      if (tok >= 0) {
        float v = (yacc[j][r] + bias) * sGate[m];
        atomicAdd(out + (size_t)tok * D_MODEL + col, v);
      }
    }
  }
}

// ---------------- launcher ----------------
extern "C" void kernel_launch(void* const* d_in, const int* in_sizes, int n_in,
                              void* d_out, int out_size, void* d_ws, size_t ws_size,
                              hipStream_t stream) {
  (void)in_sizes; (void)n_in; (void)out_size; (void)ws_size;
  const float* x  = (const float*)d_in[0];
  const float* Wr = (const float*)d_in[1];
  const float* br = (const float*)d_in[2];
  const float* W1 = (const float*)d_in[3];
  const float* b1 = (const float*)d_in[4];
  const float* W2 = (const float*)d_in[5];
  const float* b2 = (const float*)d_in[6];
  float* out = (float*)d_out;

  // workspace carve-out (~143 MB): bf16 x + L2-resident bf16 weights + routing
  uintptr_t p = (uintptr_t)d_ws;
  bf16* xb  = (bf16*)p; p += (size_t)T_TOK * D_MODEL * sizeof(bf16);
  bf16* W1t = (bf16*)p; p += (size_t)N_EXP * D_MODEL * D_FF * sizeof(bf16);
  bf16* W2t = (bf16*)p; p += (size_t)N_EXP * D_MODEL * D_FF * sizeof(bf16);
  int*   counts = (int*)p;   p += 256;
  int*   etok   = (int*)p;   p += (size_t)N_EXP * T_TOK * sizeof(int);
  float* egate  = (float*)p; p += (size_t)N_EXP * T_TOK * sizeof(float);

  hipMemsetAsync(out, 0, (size_t)T_TOK * D_MODEL * sizeof(float), stream);
  hipMemsetAsync(counts, 0, N_EXP * sizeof(int), stream);

  cvt_x_kernel<<<2048, 256, 0, stream>>>(x, xb, T_TOK * D_MODEL);
  cvt_w_tiled_kernel<<<4096, 256, 0, stream>>>(W1, W1t, D_MODEL, D_FF);
  cvt_w_tiled_kernel<<<4096, 256, 0, stream>>>(W2, W2t, D_FF, D_MODEL);
  router_kernel<<<T_TOK / 256, 256, 0, stream>>>(x, Wr, br, counts, etok, egate);

  dim3 grid(T_TOK / M_TILE, N_EXP);
  moe_ffn_kernel<<<grid, NTHREADS, 0, stream>>>(xb, W1t, W2t, b1, b2, counts, etok, egate, out);
}
